// SparseUnet_13142599926185
// MI455X (gfx1250) — matrix operand order repaired and checked by reading, not compiled, not run on hardware
//
#include <hip/hip_runtime.h>
#include <hip/hip_bf16.h>

// =====================================================================
// Sparse UNet forward for gfx1250 (MI455X).
// Core compute: gather-GEMM via V_WMMA_F32_16X16X32_F16, f32 accumulate.
// All activations kept twice: f32 [N,C] (exact residual/BN math) and
// f16 [(N+1),C] (WMMA input; row N is the zero "sentinel" row that
// rulebook sentinel indices point at).
// =====================================================================

typedef __attribute__((ext_vector_type(16))) _Float16 v16h;
typedef __attribute__((ext_vector_type(8)))  _Float16 v8h;
typedef __attribute__((ext_vector_type(8)))  float    v8f;

union ABu { v16h v; v8h h[2]; };

// ---------------------------------------------------------------------
// Pack f32 weights [27, Ci, Co] into WMMA B-fragment order (f16).
// Fragment f = ((k*cinTiles + ct)*coTiles + nt): 32 lanes x 16 halves.
// Lane holds column n = nt*16 + (lane&15), K rows ct*32+(lane>>4)*8+e (e<8), +16 for e>=8.
// Zero-pads cin >= Ci (used for the Cin=4 input conv padded to 32).
// ---------------------------------------------------------------------
__global__ void pack_w_kernel(const float* __restrict__ W, _Float16* __restrict__ Wp,
                              int Ci, int cinTiles, int coTiles, int Co) {
  int tid  = blockIdx.x * blockDim.x + threadIdx.x;
  int lane = tid & 31;
  int frag = tid >> 5;
  int nfrag = 27 * cinTiles * coTiles;
  if (frag >= nfrag) return;
  int nt   = frag % coTiles;
  int rest = frag / coTiles;
  int ct   = rest % cinTiles;
  int k    = rest / cinTiles;
  int col     = nt * 16 + (lane & 15);
  int cinBase = ct * 32 + (lane >> 4) * 8;
  ABu u;
#pragma unroll
  for (int e = 0; e < 16; ++e) {
    int cin = cinBase + (e < 8 ? e : e + 8);
    float v = (cin < Ci) ? W[((long)k * Ci + cin) * Co + col] : 0.f;
    u.v[e] = (_Float16)v;
  }
  *(v16h*)(Wp + ((long)frag << 9) + (lane << 4)) = u.v;
}

// ---------------------------------------------------------------------
// Gather-conv: Out[Nout,Co] = sum_k F16[nbr[:,k]] @ Wk  (f32 accum).
// One wave per 2x16-row M block (MT=2, amortizes each B fragment over
// two M tiles); NT 16-col accumulator tiles per wave.
// A fragment gathered straight into the ISA 16-bit A layout:
//   lane holds row (lane&15), K halves kb..kb+7 and kb+16..kb+23,
//   kb = (lane>>4)*8  -> two aligned 16B loads per 32-K chunk.
// ---------------------------------------------------------------------
template <int NT>
__global__ void __launch_bounds__(128)
conv_wmma_kernel(const _Float16* __restrict__ F, const int* __restrict__ nbr,
                 const _Float16* __restrict__ Wp, float* __restrict__ Out,
                 int Nout, int Nin, int Cip, int Co) {
  const int lane = threadIdx.x & 31;
  const int wave = threadIdx.x >> 5;
  const int m0   = (blockIdx.x * 4 + wave) * 32;   // two 16-row tiles per wave
  if (m0 >= Nout) return;                          // wave-uniform exit
  const int coTiles  = Co >> 4;
  const int cinTiles = Cip >> 5;
  const int nt0   = blockIdx.y * NT;
  const int nlane = lane & 15;
  const int half  = lane >> 4;
  const int kb    = half * 8;
  const int mrow0 = m0 + nlane;
  const int mrow1 = m0 + 16 + nlane;
  const bool mv0  = (mrow0 < Nout);
  const bool mv1  = (mrow1 < Nout);

  v8f acc0[NT] = {};
  v8f acc1[NT] = {};

  // packed-weight addressing: frag stride 512 halves; hoist bases/strides
  const _Float16* wlane = Wp + ((long)nt0 << 9) + (lane << 4);
  const long ctStride = (long)coTiles << 9;            // halves per cin-tile step
  const long kStride  = (long)cinTiles * ctStride;     // halves per k step

  for (int k = 0; k < 27; ++k) {
    int idx0 = mv0 ? nbr[(long)mrow0 * 27 + k] : Nin;  // Nin = zero row
    int idx1 = mv1 ? nbr[(long)mrow1 * 27 + k] : Nin;
    if (__ballot((idx0 != Nin) || (idx1 != Nin)) == 0) continue;  // uniform skip
    const _Float16* rowp0 = F + (long)idx0 * Cip;
    const _Float16* rowp1 = F + (long)idx1 * Cip;
    const _Float16* wk = wlane + (long)k * kStride;
    for (int ct = 0; ct < cinTiles; ++ct) {
      ABu A0, A1;
      A0.h[0] = *(const v8h*)(rowp0 + ct * 32 + kb);
      A0.h[1] = *(const v8h*)(rowp0 + ct * 32 + kb + 16);
      A1.h[0] = *(const v8h*)(rowp1 + ct * 32 + kb);
      A1.h[1] = *(const v8h*)(rowp1 + ct * 32 + kb + 16);
      const _Float16* wb = wk + ct * ctStride;
#pragma unroll
      for (int t = 0; t < NT; ++t) {
        v16h B = *(const v16h*)(wb + ((long)t << 9));
        acc0[t] = __builtin_amdgcn_wmma_f32_16x16x32_f16(
            false, A0.v, false, B, (short)0, acc0[t], false, false);
        acc1[t] = __builtin_amdgcn_wmma_f32_16x16x32_f16(
            false, A1.v, false, B, (short)0, acc1[t], false, false);
      }
    }
  }

  // C/D layout: VGPR r -> row mtile + (lane>>4)*8 + r, col = tile*16 + (lane&15)
  const int rbase = half * 8;
#pragma unroll
  for (int t = 0; t < NT; ++t) {
    int col = (nt0 + t) * 16 + nlane;
#pragma unroll
    for (int r = 0; r < 8; ++r) {
      int row = m0 + rbase + r;
      if (row < Nout) Out[(long)row * Co + col] = acc0[t][r];
      int row1 = row + 16;
      if (row1 < Nout) Out[(long)row1 * Co + col] = acc1[t][r];
    }
  }
}

// ---------------------------------------------------------------------
// Batch-norm stats, two deterministic stages (no float atomics).
// ---------------------------------------------------------------------
__global__ void colstat_partial(const float* __restrict__ X, int N, int C,
                                float* __restrict__ part) {
  __shared__ float sh[512];
  int tid = threadIdx.x;
  int c = tid % C, rep = tid / C, reps = 256 / C;
  float s = 0.f, ss = 0.f;
  for (long r = (long)blockIdx.x * reps + rep; r < N; r += (long)gridDim.x * reps) {
    float v = X[r * (long)C + c];
    s += v; ss += v * v;
  }
  sh[tid] = s; sh[256 + tid] = ss;
  __syncthreads();
  if (rep == 0) {
    for (int j = 1; j < reps; ++j) { s += sh[tid + j * C]; ss += sh[256 + tid + j * C]; }
    part[((long)blockIdx.x * C + c) * 2]     = s;
    part[((long)blockIdx.x * C + c) * 2 + 1] = ss;
  }
}

__global__ void colstat_final(const float* __restrict__ part, int NB, int N, int C,
                              float* __restrict__ stats) {
  int c = threadIdx.x;
  if (c >= C) return;
  float s = 0.f, ss = 0.f;
  for (int b = 0; b < NB; ++b) {
    s  += part[((long)b * C + c) * 2];
    ss += part[((long)b * C + c) * 2 + 1];
  }
  float inv = 1.f / (float)N;
  float m = s * inv;
  float var = ss * inv - m * m;
  stats[c * 2]     = m;
  stats[c * 2 + 1] = rsqrtf(fmaxf(var, 0.f) + 1e-3f);
}

// y = g*(x-mean)*rstd + b ; optional relu ; writes f32 and/or f16(+zero pad row)
__global__ void bn_act_kernel(const float* __restrict__ x, const float* __restrict__ stats,
                              const float* __restrict__ g, const float* __restrict__ b,
                              float* __restrict__ y32, _Float16* __restrict__ yh,
                              long N, int C, int relu) {
  long i = (long)blockIdx.x * blockDim.x + threadIdx.x;
  long total = (N + 1) * (long)C;
  if (i >= total) return;
  int c = (int)(i % C);
  long row = i / C;
  if (row < N) {
    float v = g[c] * (x[i] - stats[c * 2]) * stats[c * 2 + 1] + b[c];
    if (relu) v = fmaxf(v, 0.f);
    if (y32) y32[i] = v;
    if (yh)  yh[i] = (_Float16)v;
  } else if (yh) {
    yh[i] = (_Float16)0.f;
  }
}

// y = relu(o * (gate?gate[bidx]:1) + f) ; writes f32 + padded f16
__global__ void resid_kernel(const float* __restrict__ o, const float* __restrict__ f,
                             const float* __restrict__ gate, const int* __restrict__ bidx,
                             float* __restrict__ y32, _Float16* __restrict__ yh,
                             long N, int C) {
  long i = (long)blockIdx.x * blockDim.x + threadIdx.x;
  long total = (N + 1) * (long)C;
  if (i >= total) return;
  int c = (int)(i % C);
  long row = i / C;
  if (row < N) {
    float v = o[i];
    if (gate) v *= gate[(long)bidx[row] * C + c];
    v += f[i];
    v = fmaxf(v, 0.f);
    y32[i] = v;
    yh[i] = (_Float16)v;
  } else {
    yh[i] = (_Float16)0.f;
  }
}

// f16 pack of one or two f32 sources (concat), zero-fill pad columns + row.
__global__ void pack2_kernel(const float* __restrict__ a, int Ca,
                             const float* __restrict__ b, int Cb,
                             _Float16* __restrict__ H, long N, int Cp) {
  long i = (long)blockIdx.x * blockDim.x + threadIdx.x;
  long total = (N + 1) * (long)Cp;
  if (i >= total) return;
  int c = (int)(i % Cp);
  long row = i / Cp;
  float v = 0.f;
  if (row < N) {
    if (c < Ca) v = a[row * (long)Ca + c];
    else if (b && c < Ca + Cb) v = b[row * (long)Cb + (c - Ca)];
  }
  H[i] = (_Float16)v;
}

// SE: per-batch channel means (BSZ==2). Single block, one thread per channel.
__global__ void seg_avg_kernel(const float* __restrict__ X, const int* __restrict__ bidx,
                               int N, int C, float* __restrict__ avg) {
  int c = threadIdx.x;
  if (c >= C) return;
  float s0 = 0.f, s1 = 0.f, n0 = 0.f, n1 = 0.f;
  for (int i = 0; i < N; ++i) {
    float v = X[(long)i * C + c];
    if (bidx[i] == 0) { s0 += v; n0 += 1.f; } else { s1 += v; n1 += 1.f; }
  }
  avg[c]     = s0 / n0;
  avg[C + c] = s1 / n1;
}

// SE: gate = sigmoid(relu(avg@w1+b1)@w2+b2), per batch. One block.
__global__ void se_mlp_kernel(const float* __restrict__ avg, const float* __restrict__ w1,
                              const float* __restrict__ b1, const float* __restrict__ w2,
                              const float* __restrict__ b2, float* __restrict__ gate,
                              int C, int r) {
  __shared__ float h[64];
  int tid = threadIdx.x;
  if (tid < 2 * r) {
    int bb = tid / r, j = tid % r;
    float s = b1[j];
    for (int cc = 0; cc < C; ++cc) s += avg[bb * C + cc] * w1[cc * r + j];
    h[tid] = fmaxf(s, 0.f);
  }
  __syncthreads();
  if (tid < C) {
    for (int bb = 0; bb < 2; ++bb) {
      float s = b2[tid];
      for (int j = 0; j < r; ++j) s += h[bb * r + j] * w2[j * C + tid];
      gate[bb * C + tid] = 1.f / (1.f + expf(-s));
    }
  }
}

// =====================================================================
// Host orchestration
// =====================================================================
extern "C" void kernel_launch(void* const* d_in, const int* in_sizes, int n_in,
                              void* d_out, int out_size, void* d_ws, size_t ws_size,
                              hipStream_t stream) {
  (void)n_in; (void)out_size; (void)ws_size;

  // ---- params in JAX pytree (sorted-key) order ----
  int pi = 1;
  auto nf = [&]() -> const float* { return (const float*)d_in[pi++]; };
  struct SBBp { const float *W1,*W2,*b1,*b2,*g1,*g2,*seb1,*seb2,*sew1,*sew2; int se; };
  struct CNAp { const float *W,*b,*g; };
  auto readSBB = [&](int se) -> SBBp {
    SBBp s{}; s.W1 = nf(); s.W2 = nf(); s.b1 = nf(); s.b2 = nf();
    nf(); nf();                 // cb1, cb2: exactly cancelled by BN mean subtraction
    s.g1 = nf(); s.g2 = nf();
    s.se = se;
    if (se) { s.seb1 = nf(); s.seb2 = nf(); s.sew1 = nf(); s.sew2 = nf(); }
    return s;
  };
  auto readCNA = [&]() -> CNAp { CNAp c{}; c.W = nf(); c.b = nf(); c.g = nf(); return c; };

  SBBp c1a = readSBB(0), c1b = readSBB(0);
  SBBp c2a = readSBB(0), c2b = readSBB(0);
  CNAp d2  = readCNA();
  SBBp c3a = readSBB(0), c3b = readSBB(1);
  CNAp d3  = readCNA();
  SBBp c4a = readSBB(0), c4b = readSBB(1);
  CNAp d4  = readCNA();
  CNAp cin = readCNA();
  struct UPp { CNAp bott, out; SBBp tr; };
  auto readUP = [&]() -> UPp { UPp u; u.bott = readCNA(); u.out = readCNA(); u.tr = readSBB(0); return u; };
  UPp up1 = readUP(), up2 = readUP(), up3 = readUP(), up4 = readUP();

  const float* vf = (const float*)d_in[0];
  const int* nbr1 = (const int*)d_in[pi++];
  const int* dn2  = (const int*)d_in[pi++];
  const int* iv2  = (const int*)d_in[pi++];
  const int* nbr2 = (const int*)d_in[pi]; int i_nbr2 = pi; pi++;
  const int* dn3  = (const int*)d_in[pi++];
  const int* iv3  = (const int*)d_in[pi++];
  const int* nbr3 = (const int*)d_in[pi++];
  const int* dn4  = (const int*)d_in[pi++];
  const int* iv4  = (const int*)d_in[pi++];
  const int* nbr4 = (const int*)d_in[pi++];
  const int* b3i  = (const int*)d_in[pi]; int i_b3 = pi; pi++;
  const int* b4i  = (const int*)d_in[pi]; int i_b4 = pi; pi++;

  const long N1 = in_sizes[0] / 4;
  const long N2 = in_sizes[i_nbr2] / 27;
  const long N3 = in_sizes[i_b3];
  const long N4 = in_sizes[i_b4];

  // ---- workspace bump allocator ----
  char* base = (char*)d_ws;
  size_t off = 0;
  auto alloc = [&](size_t bytes) -> void* {
    off = (off + 255) & ~(size_t)255;
    void* p = base + off;
    off += bytes;
    return p;
  };
  auto aF = [&](long n) -> float*    { return (float*)alloc((size_t)n * 4); };
  auto aH = [&](long n) -> _Float16* { return (_Float16*)alloc((size_t)n * 2); };
  struct Feat { float* f; _Float16* h; };
  auto aFeat = [&](long N, int C) -> Feat {
    Feat x; x.f = aF(N * (long)C); x.h = aH((N + 1) * (long)C); return x;
  };

  Feat A1 = aFeat(N1, 32),  B1 = aFeat(N1, 32),  C1 = aFeat(N1, 32);
  Feat A2 = aFeat(N2, 64),  B2 = aFeat(N2, 64),  C2 = aFeat(N2, 64);
  Feat A3 = aFeat(N3, 128), B3 = aFeat(N3, 128), C3 = aFeat(N3, 128);
  Feat A4 = aFeat(N4, 256), B4 = aFeat(N4, 256), C4 = aFeat(N4, 256);
  _Float16* inH = aH((N1 + 1) * 32);
  _Float16* u2h = aH((N1 + 1) * 32);

  long maxConv = N1 * 32;
  if (N2 * 64  > maxConv) maxConv = N2 * 64;
  if (N3 * 128 > maxConv) maxConv = N3 * 128;
  if (N4 * 256 > maxConv) maxConv = N4 * 256;
  float* tmp1 = aF(maxConv);
  float* tmp2 = aF(maxConv);
  long maxTh = (N1 + 1) * 32;
  if ((N2 + 1) * 64  > maxTh) maxTh = (N2 + 1) * 64;
  if ((N3 + 1) * 128 > maxTh) maxTh = (N3 + 1) * 128;
  if ((N4 + 1) * 256 > maxTh) maxTh = (N4 + 1) * 256;
  _Float16* th = aH(maxTh);
  long maxCat = (N1 + 1) * 64;
  if ((N2 + 1) * 128 > maxCat) maxCat = (N2 + 1) * 128;
  if ((N3 + 1) * 256 > maxCat) maxCat = (N3 + 1) * 256;
  if ((N4 + 1) * 512 > maxCat) maxCat = (N4 + 1) * 512;
  _Float16* catH = aH(maxCat);
  _Float16* wpack = aH((long)27 * 16 * 16 * 512);   // largest: 27 x (512/32) x (256/16) frags
  float* part  = aF(64 * 256 * 2);
  float* stats = aF(256 * 2);
  float* avg   = aF(2 * 256);
  float* gate  = aF(2 * 256);

  auto cdiv = [](long a, long b) -> unsigned { return (unsigned)((a + b - 1) / b); };

  // ---- op launchers ----
  auto conv = [&](const _Float16* Fh, long Nin, int Cip, const int* nbr, long Nout,
                  const float* W, int Ci, int Co, float* out) {
    int cinT = Cip / 32, coT = Co / 16;
    long pthreads = (long)27 * cinT * coT * 32;
    pack_w_kernel<<<dim3(cdiv(pthreads, 256)), 256, 0, stream>>>(W, wpack, Ci, cinT, coT, Co);
    dim3 grid(cdiv(Nout, 128), (unsigned)((Co + 63) / 64));  // 4 waves x 32 rows
    if (Co == 32)
      conv_wmma_kernel<2><<<grid, 128, 0, stream>>>(Fh, nbr, wpack, out,
                                                    (int)Nout, (int)Nin, Cip, Co);
    else
      conv_wmma_kernel<4><<<grid, 128, 0, stream>>>(Fh, nbr, wpack, out,
                                                    (int)Nout, (int)Nin, Cip, Co);
  };
  auto bnstats = [&](const float* x, long N, int C) {
    colstat_partial<<<64, 256, 0, stream>>>(x, (int)N, C, part);
    colstat_final<<<1, 256, 0, stream>>>(part, 64, (int)N, C, stats);
  };
  auto bnact = [&](const float* x, const float* g, const float* b, long N, int C,
                   int relu, float* y32, _Float16* yh) {
    long total = (N + 1) * (long)C;
    bn_act_kernel<<<cdiv(total, 256), 256, 0, stream>>>(x, stats, g, b, y32, yh, N, C, relu);
  };
  auto cna = [&](const _Float16* Fh, long Nin, int Cip, const int* nbr, long Nout,
                 CNAp p, int Ci, int Co, float* y32, _Float16* yh) {
    conv(Fh, Nin, Cip, nbr, Nout, p.W, Ci, Co, tmp1);
    bnstats(tmp1, Nout, Co);
    bnact(tmp1, p.g, p.b, Nout, Co, 1, y32, yh);
  };
  auto sbb = [&](Feat in, long N, int C, const int* nbr, SBBp p, const int* bidx, Feat ob) {
    conv(in.h, N, C, nbr, N, p.W1, C, C, tmp1);
    bnstats(tmp1, N, C);
    bnact(tmp1, p.g1, p.b1, N, C, 1, nullptr, th);
    conv(th, N, C, nbr, N, p.W2, C, C, tmp1);
    bnstats(tmp1, N, C);
    bnact(tmp1, p.g2, p.b2, N, C, 0, tmp2, nullptr);
    const float* gptr = nullptr;
    if (p.se) {
      int r = C / 16;
      seg_avg_kernel<<<1, C, 0, stream>>>(tmp2, bidx, (int)N, C, avg);
      se_mlp_kernel<<<1, C, 0, stream>>>(avg, p.sew1, p.seb1, p.sew2, p.seb2, gate, C, r);
      gptr = gate;
    }
    long total = (N + 1) * (long)C;
    resid_kernel<<<cdiv(total, 256), 256, 0, stream>>>(tmp2, in.f, gptr, bidx, ob.f, ob.h, N, C);
  };
  auto pack2 = [&](const float* a, int Ca, const float* b, int Cb, _Float16* H, long N, int Cp) {
    long total = (N + 1) * (long)Cp;
    pack2_kernel<<<cdiv(total, 256), 256, 0, stream>>>(a, Ca, b, Cb, H, N, Cp);
  };

  // ---- UNet forward ----
  pack2(vf, 4, nullptr, 0, inH, N1, 32);                         // pad Cin 4 -> 32
  cna(inH, N1, 32, nbr1, N1, cin, 4, 32, A1.f, A1.h);            // f
  sbb(A1, N1, 32, nbr1, c1a, nullptr, B1);
  sbb(B1, N1, 32, nbr1, c1b, nullptr, A1);                       // f1 = A1
  cna(A1.h, N1, 32, dn2, N2, d2, 32, 64, A2.f, A2.h);
  sbb(A2, N2, 64, nbr2, c2a, nullptr, B2);
  sbb(B2, N2, 64, nbr2, c2b, nullptr, A2);                       // f2 = A2
  cna(A2.h, N2, 64, dn3, N3, d3, 64, 128, A3.f, A3.h);
  sbb(A3, N3, 128, nbr3, c3a, nullptr, B3);
  sbb(B3, N3, 128, nbr3, c3b, b3i, A3);                          // f3 = A3 (SE)
  cna(A3.h, N3, 128, dn4, N4, d4, 128, 256, A4.f, A4.h);
  sbb(A4, N4, 256, nbr4, c4a, nullptr, B4);
  sbb(B4, N4, 256, nbr4, c4b, b4i, A4);                          // f4 = A4 (SE)

  // up4 (lateral == bottom)
  sbb(A4, N4, 256, nbr4, up4.tr, nullptr, C4);                   // x
  pack2(A4.f, 256, C4.f, 256, catH, N4, 512);
  cna(catH, N4, 512, nbr4, N4, up4.bott, 512, 256, B4.f, B4.h);
  cna(B4.h, N4, 256, iv4, N3, up4.out, 256, 128, B3.f, B3.h);    // u4 = B3
  // up3
  sbb(B3, N3, 128, nbr3, up3.tr, nullptr, C3);
  pack2(A3.f, 128, C3.f, 128, catH, N3, 256);
  cna(catH, N3, 256, nbr3, N3, up3.bott, 256, 128, B3.f, B3.h);
  cna(B3.h, N3, 128, iv3, N2, up3.out, 128, 64, B2.f, B2.h);     // u3 = B2
  // up2
  sbb(B2, N2, 64, nbr2, up2.tr, nullptr, C2);
  pack2(A2.f, 64, C2.f, 64, catH, N2, 128);
  cna(catH, N2, 128, nbr2, N2, up2.bott, 128, 64, B2.f, B2.h);
  float* u2f = (float*)d_out + N1 * 32;
  cna(B2.h, N2, 64, iv2, N1, up2.out, 64, 32, u2f, u2h);         // u2 -> d_out[1]
  // up1
  Feat U2{u2f, u2h};
  sbb(U2, N1, 32, nbr1, up1.tr, nullptr, C1);
  pack2(A1.f, 32, C1.f, 32, catH, N1, 64);
  cna(catH, N1, 64, nbr1, N1, up1.bott, 64, 32, B1.f, B1.h);
  cna(B1.h, N1, 32, nbr1, N1, up1.out, 32, 32, (float*)d_out, nullptr);  // u1 -> d_out[0]
}
